// MultiHeadAttention_84215718740218
// MI455X (gfx1250) — matrix operand order repaired and checked by reading, not compiled, not run on hardware
//
#include <hip/hip_runtime.h>
#include <hip/hip_bf16.h>

// ---------------------------------------------------------------------------
// MultiHeadAttention forward for MI455X (gfx1250, wave32, WMMA).
// B=4, N=2048, E=1024, H=16, d_k=64.
// - All matmuls via v_wmma_f32_16x16x32_bf16 (fp32 accumulation).
// - Flash-attention fused softmax(QK^T)V per (b,h): 2048x2048 scores never
//   touch HBM.
// - K/V tiles streamed HBM->LDS with GLOBAL_LOAD_ASYNC_TO_LDS_B128 (ASYNCcnt),
//   double-buffered so the DMA of block j+1 overlaps the 32 WMMAs of block j.
// ---------------------------------------------------------------------------

#define E_DIM  1024
#define HEADS  16
#define DK     64
#define NSEQ   2048
#define BATCH  4
#define MTOT   (BATCH * NSEQ)   // 8192 tokens total

typedef __bf16 bf16_t;
typedef __attribute__((ext_vector_type(16))) __bf16 bf16x16;
typedef __attribute__((ext_vector_type(8)))  __bf16 bf16x8;
typedef __attribute__((ext_vector_type(8)))  float  f32x8;

static __device__ __forceinline__ f32x8 wmma_bf16(bf16x16 a, bf16x16 b, f32x8 c) {
  // 8-arg form: (neg_a, A, neg_b, B, c_mod, C, reuse_a, reuse_b)
  return __builtin_amdgcn_wmma_f32_16x16x32_bf16(false, a, false, b, (short)0, c,
                                                 false, false);
}

static __device__ __forceinline__ bf16x16 join8(bf16x8 lo, bf16x8 hi) {
  return __builtin_shufflevector(lo, hi, 0, 1, 2, 3, 4, 5, 6, 7,
                                 8, 9, 10, 11, 12, 13, 14, 15);
}

// Async DMA: 16 bytes global -> LDS, tracked by ASYNCcnt (no VGPR round-trip).
// LDS byte address = low 32 bits of the flat address (aperture rule).
static __device__ __forceinline__ void async_b128(const bf16_t* gsrc, bf16_t* ldst) {
  uint32_t lds = (uint32_t)(uintptr_t)ldst;
  asm volatile("global_load_async_to_lds_b128 %0, %1, off"
               :: "v"(lds), "v"(gsrc) : "memory");
}
static __device__ __forceinline__ void wait_async_le8() {
  asm volatile("s_wait_asynccnt 0x8" ::: "memory");
}
static __device__ __forceinline__ void wait_async_0() {
  asm volatile("s_wait_asynccnt 0x0" ::: "memory");
}

// ---------------------------------------------------------------------------
// Kernel 1: W[in][out] fp32 -> Wt[out][in] bf16 (tiled transpose through LDS).
// ---------------------------------------------------------------------------
__global__ __launch_bounds__(256) void wt_transpose_kernel(
    const float* __restrict__ Wq, const float* __restrict__ Wk,
    const float* __restrict__ Wv, const float* __restrict__ Wo,
    bf16_t* __restrict__ wt_all) {
  __shared__ float tile[16][17];
  const int sel = blockIdx.z;
  const float* src = (sel == 0) ? Wq : (sel == 1) ? Wk : (sel == 2) ? Wv : Wo;
  bf16_t* dst = wt_all + (size_t)sel * E_DIM * E_DIM;
  const int n0 = blockIdx.x * 16, k0 = blockIdx.y * 16;
  const int tx = threadIdx.x, ty = threadIdx.y;
  tile[ty][tx] = src[(size_t)(k0 + ty) * E_DIM + (n0 + tx)];
  __syncthreads();
  dst[(size_t)(n0 + ty) * E_DIM + (k0 + tx)] = (bf16_t)tile[tx][ty];
}

// ---------------------------------------------------------------------------
// Kernel 2: fused QKV projection. Y = x @ W_sel + b_sel, sel = blockIdx.z.
//   sel 0: Q, scaled by 1/sqrt(dk)=0.125, layout [b][h][token][dk]
//   sel 1: K,                              layout [b][h][token][dk]
//   sel 2: V, written transposed           layout [b][h][dk][token]
// Block tile 128x128, 8 waves (4x2), each wave 32x64 = 2x4 WMMA tiles.
// ---------------------------------------------------------------------------
__global__ __launch_bounds__(256) void qkv_gemm_kernel(
    const float* __restrict__ x, const bf16_t* __restrict__ wt_all,
    const float* __restrict__ bq, const float* __restrict__ bk,
    const float* __restrict__ bv,
    bf16_t* __restrict__ Q, bf16_t* __restrict__ K, bf16_t* __restrict__ Vt) {
  __shared__ alignas(32) bf16_t As[128][32];   // [m][k], k contiguous
  __shared__ alignas(32) bf16_t Bs[128][32];   // [n][k], k contiguous

  const int sel = blockIdx.z;
  const bf16_t* wt = wt_all + (size_t)sel * E_DIM * E_DIM;
  const float* bias = (sel == 0) ? bq : (sel == 1) ? bk : bv;
  const int m0 = blockIdx.y * 128, n0 = blockIdx.x * 128;
  const int tid = threadIdx.x;
  const int lane = tid & 31, wave = tid >> 5;
  const int wm = wave >> 1, wn = wave & 1;      // 4x2 wave grid

  const f32x8 fz = {0.f, 0.f, 0.f, 0.f, 0.f, 0.f, 0.f, 0.f};
  f32x8 acc[2][4];
#pragma unroll
  for (int mt = 0; mt < 2; ++mt)
#pragma unroll
    for (int nt = 0; nt < 4; ++nt) acc[mt][nt] = fz;

  const int rowL = tid >> 1, halfL = tid & 1;   // cooperative tile loads
  const int arow = lane & 15;
  const int akb = (lane < 16) ? 0 : 16;         // contiguous A-layout K base
  const int bkb = (lane < 16) ? 0 : 16;         // contiguous B-layout K base

  for (int k0 = 0; k0 < E_DIM; k0 += 32) {
    __syncthreads();
    {  // B tile: async DMA straight into LDS (ASYNCcnt path)
      const bf16_t* src = wt + (size_t)(n0 + rowL) * E_DIM + k0 + halfL * 16;
      bf16_t* dst = &Bs[rowL][halfL * 16];
      async_b128(src, dst);
      async_b128(src + 8, dst + 8);
    }
    {  // A tile: x fp32 -> bf16 through VALU (needs conversion)
      const float4* src =
          (const float4*)(x + (size_t)(m0 + rowL) * E_DIM + k0 + halfL * 16);
      bf16_t* dst = &As[rowL][halfL * 16];
#pragma unroll
      for (int i = 0; i < 4; ++i) {
        float4 v = src[i];
        dst[i * 4 + 0] = (bf16_t)v.x; dst[i * 4 + 1] = (bf16_t)v.y;
        dst[i * 4 + 2] = (bf16_t)v.z; dst[i * 4 + 3] = (bf16_t)v.w;
      }
    }
    if (k0 + 32 < E_DIM) {  // prefetch next x tile into cache
      __builtin_prefetch(x + (size_t)(m0 + rowL) * E_DIM + k0 + 32 + halfL * 16, 0, 1);
    }
    wait_async_0();
    __syncthreads();

    bf16x16 afrag[2];
#pragma unroll
    for (int mt = 0; mt < 2; ++mt) {
      const bf16_t* ap = &As[wm * 32 + mt * 16 + arow][0];
      afrag[mt] = join8(*(const bf16x8*)(ap + akb), *(const bf16x8*)(ap + akb + 8));
    }
#pragma unroll
    for (int nt = 0; nt < 4; ++nt) {
      bf16x16 bfrag = *(const bf16x16*)&Bs[wn * 64 + nt * 16 + arow][bkb];
#pragma unroll
      for (int mt = 0; mt < 2; ++mt)
        acc[mt][nt] = wmma_bf16(afrag[mt], bfrag, acc[mt][nt]);
    }
  }

  // Epilogue: bias add, cast, scatter into per-head layouts.
  const int mh = (lane < 16) ? 0 : 8;
#pragma unroll
  for (int mt = 0; mt < 2; ++mt)
#pragma unroll
    for (int nt = 0; nt < 4; ++nt)
#pragma unroll
      for (int r = 0; r < 8; ++r) {
        const int m = m0 + wm * 32 + mt * 16 + r + mh;        // token row
        const int n = n0 + wn * 64 + nt * 16 + (lane & 15);   // feature col
        float v = acc[mt][nt][r] + bias[n];
        const int b = m >> 11, token = m & (NSEQ - 1);
        const int h = n >> 6, dk = n & 63;
        const size_t bh = (size_t)b * HEADS + h;
        if (sel == 0) {
          Q[(bh * NSEQ + token) * DK + dk] = (bf16_t)(v * 0.125f);  // fold 1/sqrt(dk)
        } else if (sel == 1) {
          K[(bh * NSEQ + token) * DK + dk] = (bf16_t)v;
        } else {
          Vt[(bh * DK + dk) * NSEQ + token] = (bf16_t)v;            // transposed V
        }
      }
}

// ---------------------------------------------------------------------------
// Kernel 3: flash attention core. One block per (b*h, 128-query block).
// 8 waves; each wave owns 16 query rows. K/V processed in 128-key blocks,
// double-buffered via async DMA to LDS; online softmax; 32 WMMAs/wave/block.
// ---------------------------------------------------------------------------
__global__ __launch_bounds__(256) void attn_kernel(
    const bf16_t* __restrict__ Q, const bf16_t* __restrict__ K,
    const bf16_t* __restrict__ Vt, bf16_t* __restrict__ attn) {
  __shared__ alignas(32) bf16_t Ks[2][128][64];    // [buf][key][dk]
  __shared__ alignas(32) bf16_t Vts[2][64][128];   // [buf][dv][key]
  __shared__ alignas(32) bf16_t Ps[8][16][128];    // per-wave P staging

  const int bh = blockIdx.y;
  const int q0 = blockIdx.x * 128;
  const int tid = threadIdx.x, lane = tid & 31, wave = tid >> 5;
  const bf16_t* Kb = K + (size_t)bh * NSEQ * DK;
  const bf16_t* Vb = Vt + (size_t)bh * DK * NSEQ;

  const int arow = lane & 15;
  const int akb = (lane < 16) ? 0 : 16;
  const int bkb = (lane < 16) ? 0 : 16;

  // Async K/V tile fetch for key block starting at j0 into buffer `buf`:
  // 8 x b128 per thread (16KB K + 16KB V per block), per-lane LDS addresses.
  auto issue_kv = [&](int j0, int buf) {
    const int key = tid >> 1, kh = tid & 1;
    const bf16_t* ksrc = Kb + (size_t)(j0 + key) * DK + kh * 32;
    bf16_t* kdst = &Ks[buf][key][kh * 32];
    const int dv = tid >> 2, qrt = tid & 3;
    const bf16_t* vsrc = Vb + (size_t)dv * NSEQ + j0 + qrt * 32;
    bf16_t* vdst = &Vts[buf][dv][qrt * 32];
#pragma unroll
    for (int i = 0; i < 4; ++i) {
      async_b128(ksrc + i * 8, kdst + i * 8);
      async_b128(vsrc + i * 8, vdst + i * 8);
    }
  };

  // Q A-fragments, loaded once straight from global (row-major [token][dk]).
  const bf16_t* qp = Q + ((size_t)bh * NSEQ + q0 + wave * 16 + arow) * DK;
  const bf16x16 qa0 = join8(*(const bf16x8*)(qp + akb),
                            *(const bf16x8*)(qp + akb + 8));
  const bf16x16 qa1 = join8(*(const bf16x8*)(qp + 32 + akb),
                            *(const bf16x8*)(qp + 32 + akb + 8));

  const f32x8 fz = {0.f, 0.f, 0.f, 0.f, 0.f, 0.f, 0.f, 0.f};
  f32x8 o[4];
  float mrow[8], lrow[8];
#pragma unroll
  for (int i = 0; i < 4; ++i) o[i] = fz;
#pragma unroll
  for (int r = 0; r < 8; ++r) { mrow[r] = -1e30f; lrow[r] = 0.f; }

  issue_kv(0, 0);  // prime the pipeline

  const int NBLK = NSEQ / 128;
  for (int jb = 0; jb < NBLK; ++jb) {
    const int cur = jb & 1;
    const bool has_next = (jb + 1) < NBLK;
    if (has_next) {
      issue_kv((jb + 1) * 128, cur ^ 1);  // DMA next block while computing
      wait_async_le8();                   // current block's 8 DMAs complete
    } else {
      wait_async_0();
    }
    __syncthreads();  // all waves' DMA for buf[cur] visible

    // ---- scores: S (16 x 128) = Qs (16x64) * K^T ----
    f32x8 s[8];
#pragma unroll
    for (int nt = 0; nt < 8; ++nt) {
      const bf16_t* kp = &Ks[cur][nt * 16 + arow][0];
      bf16x16 b0 = *(const bf16x16*)(kp + bkb);
      bf16x16 b1 = *(const bf16x16*)(kp + 32 + bkb);
      f32x8 a = fz;
      a = wmma_bf16(qa0, b0, a);
      a = wmma_bf16(qa1, b1, a);
      s[nt] = a;
    }

    // ---- online softmax (row r lives in VGPR r across a 16-lane group) ----
    float lm[8];
#pragma unroll
    for (int r = 0; r < 8; ++r) {
      float v = s[0][r];
#pragma unroll
      for (int nt = 1; nt < 8; ++nt) v = fmaxf(v, s[nt][r]);
#pragma unroll
      for (int off = 1; off < 16; off <<= 1)
        v = fmaxf(v, __shfl_xor(v, off, 32));
      lm[r] = v;
    }
#pragma unroll
    for (int r = 0; r < 8; ++r) {
      float mnew = fmaxf(mrow[r], lm[r]);
      float fac = __expf(mrow[r] - mnew);
      mrow[r] = mnew;
      lrow[r] *= fac;
#pragma unroll
      for (int nt = 0; nt < 4; ++nt) o[nt][r] *= fac;
    }
    float rs[8];
#pragma unroll
    for (int r = 0; r < 8; ++r) rs[r] = 0.f;
#pragma unroll
    for (int nt = 0; nt < 8; ++nt)
#pragma unroll
      for (int r = 0; r < 8; ++r) {
        float p = __expf(s[nt][r] - mrow[r]);
        s[nt][r] = p;
        rs[r] += p;
      }
#pragma unroll
    for (int r = 0; r < 8; ++r) {
      float v = rs[r];
#pragma unroll
      for (int off = 1; off < 16; off <<= 1) v += __shfl_xor(v, off, 32);
      lrow[r] += v;
    }

    // ---- relayout P (D layout -> A layout) via per-wave LDS ----
    const int mh = (lane < 16) ? 0 : 8;
#pragma unroll
    for (int nt = 0; nt < 8; ++nt)
#pragma unroll
      for (int r = 0; r < 8; ++r)
        Ps[wave][r + mh][nt * 16 + arow] = (bf16_t)s[nt][r];

    // ---- O += P (16x128) * V (128x64) ----
#pragma unroll
    for (int kt = 0; kt < 4; ++kt) {
      const bf16_t* pp = &Ps[wave][arow][kt * 32];
      bf16x16 a = join8(*(const bf16x8*)(pp + akb), *(const bf16x8*)(pp + akb + 8));
#pragma unroll
      for (int nt = 0; nt < 4; ++nt) {
        bf16x16 bv = *(const bf16x16*)&Vts[cur][nt * 16 + arow][kt * 32 + bkb];
        o[nt] = wmma_bf16(a, bv, o[nt]);
      }
    }
    __syncthreads();  // all waves done with buf[cur] before it is re-filled
  }

  // ---- normalize and write attn [b][token][h*64+dv] as bf16 ----
  const int b = bh / HEADS, h = bh % HEADS;
  const int mh = (lane < 16) ? 0 : 8;
#pragma unroll
  for (int nt = 0; nt < 4; ++nt)
#pragma unroll
    for (int r = 0; r < 8; ++r) {
      const int token = q0 + wave * 16 + r + mh;
      const int dv = nt * 16 + arow;
      float v = o[nt][r] / lrow[r];
      attn[((size_t)b * NSEQ + token) * E_DIM + h * DK + dv] = (bf16_t)v;
    }
}

// ---------------------------------------------------------------------------
// Kernel 4: output projection. out = attn(bf16) @ Wo + b_o, fp32 result.
// ---------------------------------------------------------------------------
__global__ __launch_bounds__(256) void oproj_gemm_kernel(
    const bf16_t* __restrict__ Ain, const bf16_t* __restrict__ wt,
    const float* __restrict__ bias, float* __restrict__ out) {
  __shared__ alignas(32) bf16_t As[128][32];
  __shared__ alignas(32) bf16_t Bs[128][32];

  const int m0 = blockIdx.y * 128, n0 = blockIdx.x * 128;
  const int tid = threadIdx.x;
  const int lane = tid & 31, wave = tid >> 5;
  const int wm = wave >> 1, wn = wave & 1;

  const f32x8 fz = {0.f, 0.f, 0.f, 0.f, 0.f, 0.f, 0.f, 0.f};
  f32x8 acc[2][4];
#pragma unroll
  for (int mt = 0; mt < 2; ++mt)
#pragma unroll
    for (int nt = 0; nt < 4; ++nt) acc[mt][nt] = fz;

  const int rowL = tid >> 1, halfL = tid & 1;
  const int arow = lane & 15;
  const int akb = (lane < 16) ? 0 : 16;
  const int bkb = (lane < 16) ? 0 : 16;

  for (int k0 = 0; k0 < E_DIM; k0 += 32) {
    __syncthreads();
    {  // A tile (already bf16): async DMA to LDS
      const bf16_t* src = Ain + (size_t)(m0 + rowL) * E_DIM + k0 + halfL * 16;
      bf16_t* dst = &As[rowL][halfL * 16];
      async_b128(src, dst);
      async_b128(src + 8, dst + 8);
    }
    {  // B tile: async DMA to LDS
      const bf16_t* src = wt + (size_t)(n0 + rowL) * E_DIM + k0 + halfL * 16;
      bf16_t* dst = &Bs[rowL][halfL * 16];
      async_b128(src, dst);
      async_b128(src + 8, dst + 8);
    }
    wait_async_0();
    __syncthreads();

    bf16x16 afrag[2];
#pragma unroll
    for (int mt = 0; mt < 2; ++mt) {
      const bf16_t* ap = &As[wm * 32 + mt * 16 + arow][0];
      afrag[mt] = join8(*(const bf16x8*)(ap + akb), *(const bf16x8*)(ap + akb + 8));
    }
#pragma unroll
    for (int nt = 0; nt < 4; ++nt) {
      bf16x16 bfrag = *(const bf16x16*)&Bs[wn * 64 + nt * 16 + arow][bkb];
#pragma unroll
      for (int mt = 0; mt < 2; ++mt)
        acc[mt][nt] = wmma_bf16(afrag[mt], bfrag, acc[mt][nt]);
    }
  }

  const int mh = (lane < 16) ? 0 : 8;
#pragma unroll
  for (int mt = 0; mt < 2; ++mt)
#pragma unroll
    for (int nt = 0; nt < 4; ++nt)
#pragma unroll
      for (int r = 0; r < 8; ++r) {
        const int m = m0 + wm * 32 + mt * 16 + r + mh;
        const int n = n0 + wn * 64 + nt * 16 + (lane & 15);
        out[(size_t)m * E_DIM + n] = acc[mt][nt][r] + bias[n];
      }
}

// ---------------------------------------------------------------------------
extern "C" void kernel_launch(void* const* d_in, const int* in_sizes, int n_in,
                              void* d_out, int out_size, void* d_ws, size_t ws_size,
                              hipStream_t stream) {
  (void)in_sizes; (void)n_in; (void)out_size; (void)ws_size;

  const float* x  = (const float*)d_in[0];
  const float* Wq = (const float*)d_in[1];
  const float* bq = (const float*)d_in[2];
  const float* Wk = (const float*)d_in[3];
  const float* bk = (const float*)d_in[4];
  const float* Wv = (const float*)d_in[5];
  const float* bv = (const float*)d_in[6];
  const float* Wo = (const float*)d_in[7];
  const float* bo = (const float*)d_in[8];
  float* out = (float*)d_out;

  // Workspace layout (bf16 elements): 4 transposed weights, Q, K, V^T, attn.
  bf16_t* ws = (bf16_t*)d_ws;
  const size_t EE  = (size_t)E_DIM * E_DIM;              // 1 Mi elems
  const size_t QKV = (size_t)BATCH * HEADS * NSEQ * DK;  // 8 Mi elems
  bf16_t* wt_all = ws;              // 4*EE
  bf16_t* Qb   = ws + 4 * EE;
  bf16_t* Kb   = Qb + QKV;
  bf16_t* Vtb  = Kb + QKV;
  bf16_t* attn = Vtb + QKV;         // total ~72 MB

  wt_transpose_kernel<<<dim3(E_DIM / 16, E_DIM / 16, 4), dim3(16, 16), 0, stream>>>(
      Wq, Wk, Wv, Wo, wt_all);
  qkv_gemm_kernel<<<dim3(E_DIM / 128, MTOT / 128, 3), 256, 0, stream>>>(
      x, wt_all, bq, bk, bv, Qb, Kb, Vtb);
  attn_kernel<<<dim3(NSEQ / 128, BATCH * HEADS), 256, 0, stream>>>(
      Qb, Kb, Vtb, attn);
  oproj_gemm_kernel<<<dim3(E_DIM / 128, MTOT / 128), 256, 0, stream>>>(
      attn, wt_all + 3 * EE, bo, out);
}
